// Attention_12713103197090
// MI455X (gfx1250) — matrix operand order repaired and checked by reading, not compiled, not run on hardware
//
#include <hip/hip_runtime.h>
#include <hip/hip_bf16.h>
#include <stdint.h>

// ---------------------------------------------------------------------------
// GMM-attention LSTM (B=32,T=512,S=1024,H=512,M=16) as ONE persistent kernel.
// 32 workgroups x 256 threads (8 wave32 waves) = 256 waves == 256 WMMA tiles
// of the per-step gate GEMM  gates = [x_t|ctx|h](32x1536) @ Wcat^T(1536x2048).
// Weights pre-packed to bf16 WMMA B-fragments (L2-resident, 6MB).
// Activations packed to bf16 A-fragments each step, staged into LDS with
// double-buffered async-to-LDS copies. fp32 accumulate everywhere.
// ---------------------------------------------------------------------------

typedef __attribute__((ext_vector_type(16))) __bf16 v16bf;
typedef __attribute__((ext_vector_type(8)))  float  v8f;

// async-to-LDS builtin pointer types: (v4i AS1* src, v4i AS3* dst, imm, imm)
typedef int v4i_ __attribute__((vector_size(16)));
typedef __attribute__((address_space(1))) v4i_* gv4i;
typedef __attribute__((address_space(3))) v4i_* lv4i;

#define NBLK   32
#define NTHR   256
#define GSZ    (NBLK * NTHR)
#define Bb     32
#define Tt     512
#define Ss     1024
#define Hh     512
#define Mm     16
#define KTOT   1536          // 3*H
#define NG     2048          // 4*H
#define NKT    48            // KTOT/32
#define CH     8             // K-tiles per LDS chunk
#define NCHUNK (NKT / CH)

// ---- workspace byte offsets ----
#define WS_BAR   0ull
#define WS_WB    256ull                       // 128*48*512 ushort = 6 MB
#define WS_BIAS  (WS_WB    + 6291456ull)      // 2048 f32
#define WS_XB    (WS_BIAS  + 8192ull)         // 2*48*512 ushort = 96 KB
#define WS_GATES (WS_XB    + 98304ull)        // 32*2048 f32
#define WS_HS    (WS_GATES + 262144ull)       // 32*512 f32
#define WS_CS    (WS_HS    + 65536ull)
#define WS_CTX   (WS_CS    + 65536ull)
#define WS_KSI   (WS_CTX   + 65536ull)        // 32*16 f32
#define WS_BETA  (WS_KSI   + 2048ull)
#define WS_ALPHA (WS_BETA  + 2048ull)
#define WS_PHI   (WS_ALPHA + 2048ull)         // 32*48 f32
#define WS_WV    (WS_PHI   + 6144ull)         // 32*1024 f32

__device__ __forceinline__ unsigned short f2bf(float f) {
  union { float f; unsigned u; } v; v.f = f;
  unsigned r = v.u + 0x7FFFu + ((v.u >> 16) & 1u);   // round-to-nearest-even
  return (unsigned short)(r >> 16);
}
__device__ __forceinline__ float sigm(float x) { return 1.0f / (1.0f + __expf(-x)); }

// 16-bit A-matrix 16x32 lane layout (ISA 7.12.2): lanes 0-15 hold K 0..7,16..23;
// lanes 16-31 hold K 8..15,24..31 (pairs packed per dword).
__device__ __forceinline__ int a_klocal(int lane, int j) {
  int v = j >> 1, hb = j & 1;
  int base = (v < 4) ? (v * 2 + hb) : (16 + (v - 4) * 2 + hb);
  return (lane < 16) ? base : base + 8;
}

__device__ __forceinline__ void async_cp16(const void* g, void* l) {
#if __has_builtin(__builtin_amdgcn_global_load_async_to_lds_b128)
  __builtin_amdgcn_global_load_async_to_lds_b128(
      (gv4i)(uintptr_t)g,
      (lv4i)(unsigned)(uintptr_t)l,
      0, 0);
#else
  *(uint4*)l = *(const uint4*)g;   // synchronous fallback
#endif
}

__device__ __forceinline__ void wait_async0() {
#if __has_builtin(__builtin_amdgcn_s_wait_asynccnt)
  __builtin_amdgcn_s_wait_asynccnt(0);
#else
  asm volatile("s_wait_asynccnt 0" ::: "memory");
#endif
}

__device__ __forceinline__ void grid_barrier(unsigned* bar) {
  __builtin_amdgcn_fence(__ATOMIC_RELEASE, "agent");
  __syncthreads();
  if (threadIdx.x == 0) {
    unsigned* cnt = bar;
    unsigned* gen = bar + 1;
    unsigned g = __atomic_load_n(gen, __ATOMIC_ACQUIRE);
    if (atomicAdd(cnt, 1u) == (unsigned)(NBLK - 1)) {
      __atomic_store_n(cnt, 0u, __ATOMIC_RELAXED);
      __atomic_store_n(gen, g + 1u, __ATOMIC_RELEASE);
    } else {
      while (__atomic_load_n(gen, __ATOMIC_ACQUIRE) == g)
        __builtin_amdgcn_s_sleep(1);
    }
  }
  __syncthreads();
  __builtin_amdgcn_fence(__ATOMIC_ACQUIRE, "agent");
}

extern "C" __global__ __launch_bounds__(NTHR, 2)
void gmm_attn_lstm(const float* __restrict__ xin,  const float* __restrict__ mem,
                   const float* __restrict__ Wih,  const float* __restrict__ Whh,
                   const float* __restrict__ bih,  const float* __restrict__ bhh,
                   const float* __restrict__ Wg,   const float* __restrict__ bgv,
                   float* __restrict__ out_ctx,    float* __restrict__ out_align,
                   char* __restrict__ ws)
{
  unsigned*       bar   = (unsigned*)(ws + WS_BAR);
  unsigned short* gWb   = (unsigned short*)(ws + WS_WB);
  float*          bias  = (float*)(ws + WS_BIAS);
  unsigned short* gXb   = (unsigned short*)(ws + WS_XB);
  float*          gates = (float*)(ws + WS_GATES);
  float*          hS    = (float*)(ws + WS_HS);
  float*          cS    = (float*)(ws + WS_CS);
  float*          ctxS  = (float*)(ws + WS_CTX);
  float*          ksi   = (float*)(ws + WS_KSI);
  float*          beta  = (float*)(ws + WS_BETA);
  float*          alpha = (float*)(ws + WS_ALPHA);
  float*          phi   = (float*)(ws + WS_PHI);
  float*          wV    = (float*)(ws + WS_WV);

  const int tid  = threadIdx.x;
  const int gtid = blockIdx.x * NTHR + tid;
  const int lane = tid & 31;
  const int wgl  = blockIdx.x * (NTHR / 32) + (tid >> 5);  // global wave id 0..255
  const int wmt  = wgl & 1;                                // M tile (0..1)
  const int wnt  = wgl >> 1;                               // N tile (0..127)

  __shared__ __attribute__((aligned(32))) unsigned short sA[2][2 * CH * 512];
  __shared__ float sw[Ss];

  // ---- prep: pack Wcat = [W_ih ; W_hh] into bf16 WMMA B fragments + bias; zero state
  for (int f = gtid; f < 128 * NKT * 512; f += GSZ) {
    int nt = f / (NKT * 512);
    int r0 = f - nt * (NKT * 512);
    int kt = r0 >> 9;
    int l  = r0 & 511;
    int ln = l >> 4, j = l & 15;
    int n  = nt * 16 + (ln & 15);
    int k  = kt * 32 + ((j < 8) ? j : j + 8) + ((ln < 16) ? 0 : 8);   // B 32x16 lane layout
    float v = (k < 2 * Hh) ? Wih[(size_t)n * (2 * Hh) + k]
                           : Whh[(size_t)n * Hh + (k - 2 * Hh)];
    gWb[f] = f2bf(v);
  }
  for (int n = gtid; n < NG; n += GSZ) bias[n] = bih[n] + bhh[n];
  for (int i = gtid; i < Bb * Hh; i += GSZ) { hS[i] = 0.f; cS[i] = 0.f; ctxS[i] = 0.f; }
  for (int i = gtid; i < Bb * Mm; i += GSZ) ksi[i] = 0.f;
  grid_barrier(bar);

  for (int t = 0; t < Tt; ++t) {
    // ---- phase 1: build X = [x_t | ctx | h] as bf16 A fragments
    for (int f = gtid; f < 2 * NKT * 512; f += GSZ) {
      int mt = f / (NKT * 512);
      int r0 = f - mt * (NKT * 512);
      int kt = r0 >> 9;
      int l  = r0 & 511;
      int ln = l >> 4, j = l & 15;
      int m  = mt * 16 + (ln & 15);
      int k  = kt * 32 + a_klocal(ln, j);
      float v;
      if (k < Hh)          v = xin[((size_t)m * Tt + t) * Hh + k];
      else if (k < 2 * Hh) v = ctxS[m * Hh + (k - Hh)];
      else                 v = hS[m * Hh + (k - 2 * Hh)];
      gXb[f] = f2bf(v);
    }
    grid_barrier(bar);

    // ---- phase 2: gate GEMM via bf16 WMMA, async-to-LDS double-buffered A
    {
      v8f acc;
      float bsv = bias[wnt * 16 + (lane & 15)];
      #pragma unroll
      for (int r = 0; r < 8; ++r) acc[r] = bsv;

      for (int u = tid; u < 2 * CH * 64; u += NTHR) {       // chunk 0 (16 KB)
        int mt2 = u >> 9, rem = u & 511, ktl = rem >> 6, e8 = rem & 63;
        async_cp16(gXb + (((size_t)mt2 * NKT + ktl) * 512 + e8 * 8),
                   &sA[0][(mt2 * CH + ktl) * 512 + e8 * 8]);
      }
      for (int c = 0; c < NCHUNK; ++c) {
        wait_async0();
        __syncthreads();
        if (c + 1 < NCHUNK) {                               // prefetch next chunk
          int kt0 = (c + 1) * CH;
          for (int u = tid; u < 2 * CH * 64; u += NTHR) {
            int mt2 = u >> 9, rem = u & 511, ktl = rem >> 6, e8 = rem & 63;
            async_cp16(gXb + (((size_t)mt2 * NKT + kt0 + ktl) * 512 + e8 * 8),
                       &sA[(c + 1) & 1][(mt2 * CH + ktl) * 512 + e8 * 8]);
          }
        }
        #pragma unroll
        for (int ktl = 0; ktl < CH; ++ktl) {
          int kt = c * CH + ktl;
          v16bf a = *(const v16bf*)&sA[c & 1][(wmt * CH + ktl) * 512 + lane * 16];
          v16bf b = *(const v16bf*)&gWb[((size_t)wnt * NKT + kt) * 512 + lane * 16];
          acc = __builtin_amdgcn_wmma_f32_16x16x32_bf16(false, a, false, b,
                                                        (short)0, acc, false, false);
        }
      }
      int nc = wnt * 16 + (lane & 15);
      #pragma unroll
      for (int r = 0; r < 8; ++r) {
        int m = wmt * 16 + r + ((lane < 16) ? 0 : 8);       // C/D lane layout
        gates[(size_t)m * NG + nc] = acc[r];
      }
    }
    grid_barrier(bar);

    // ---- phase 3: LSTM cell
    for (int i = gtid; i < Bb * Hh; i += GSZ) {
      int b = i >> 9, d = i & (Hh - 1);
      const float* g = gates + (size_t)b * NG;
      float ig = sigm(g[d]);
      float fg = sigm(g[Hh + d]);
      float gg = tanhf(g[2 * Hh + d]);
      float og = sigm(g[3 * Hh + d]);
      float cn = fg * cS[i] + ig * gg;
      cS[i] = cn;
      hS[i] = og * tanhf(cn);
    }
    grid_barrier(bar);

    // ---- phase 4: phi = h @ Wg^T + bg   (32x48, K=512 — VALU dot products)
    if (gtid < Bb * 3 * Mm) {
      int b = gtid / (3 * Mm), j = gtid % (3 * Mm);
      float a0 = bgv[j];
      const float* hr = hS + (size_t)b * Hh;
      const float* wr = Wg + (size_t)j * Hh;
      for (int d = 0; d < Hh; ++d) a0 += hr[d] * wr[d];
      phi[b * 3 * Mm + j] = a0;
    }
    grid_barrier(bar);

    // ---- phase 5: mixture params: ksi += exp, beta = exp, alpha = softmax
    if (blockIdx.x == 0 && tid < Bb) {
      int b = tid;
      const float* p = phi + b * 3 * Mm;
      float mx = -3.4e38f;
      float av[Mm];
      #pragma unroll
      for (int m = 0; m < Mm; ++m) { av[m] = p[2 * Mm + m]; mx = fmaxf(mx, av[m]); }
      float se = 0.f;
      #pragma unroll
      for (int m = 0; m < Mm; ++m) { av[m] = __expf(av[m] - mx); se += av[m]; }
      float inv = 1.f / se;
      #pragma unroll
      for (int m = 0; m < Mm; ++m) {
        alpha[b * Mm + m] = av[m] * inv;
        ksi[b * Mm + m]  += __expf(p[m]);
        beta[b * Mm + m]  = __expf(p[Mm + m]);
      }
    }
    grid_barrier(bar);

    // ---- phase 6: alignment weights w[b][s] (and output)
    for (int i = gtid; i < Bb * Ss; i += GSZ) {
      int b = i >> 10, s = i & (Ss - 1);
      float uR = (float)s + 1.5f, uL = (float)s + 0.5f;
      float a0 = 0.f;
      #pragma unroll
      for (int m = 0; m < Mm; ++m) {
        float ks = ksi[b * Mm + m];
        float ib = 1.f / beta[b * Mm + m];
        a0 += alpha[b * Mm + m] * (sigm((uR - ks) * ib) - sigm((uL - ks) * ib));
      }
      wV[i] = a0;
      out_align[((size_t)b * Tt + t) * Ss + s] = a0;
    }
    grid_barrier(bar);

    // ---- phase 7: ctx = w @ memory (one batch per WG, w in LDS, coalesced mem)
    {
      int b = blockIdx.x;
      for (int i = tid; i < Ss; i += NTHR) sw[i] = wV[b * Ss + i];
      __syncthreads();
      for (int d = tid; d < Hh; d += NTHR) {
        const float* mb = mem + (size_t)b * Ss * Hh + d;
        float a0 = 0.f;
        for (int s = 0; s < Ss; ++s) {
          if ((s & 63) == 0) __builtin_prefetch(mb + (size_t)(s + 64) * Hh, 0, 0);
          a0 += sw[s] * mb[(size_t)s * Hh];
        }
        ctxS[b * Hh + d] = a0;
        out_ctx[((size_t)b * Tt + t) * Hh + d] =
            a0 + xin[((size_t)b * Tt + t) * Hh + d];
      }
      __syncthreads();
    }
    grid_barrier(bar);
  }
}

extern "C" void kernel_launch(void* const* d_in, const int* in_sizes, int n_in,
                              void* d_out, int out_size, void* d_ws, size_t ws_size,
                              hipStream_t stream) {
  (void)in_sizes; (void)n_in; (void)out_size; (void)ws_size;
  const float* xin = (const float*)d_in[0];
  const float* mem = (const float*)d_in[1];
  const float* Wih = (const float*)d_in[2];
  const float* Whh = (const float*)d_in[3];
  const float* bih = (const float*)d_in[4];
  const float* bhh = (const float*)d_in[5];
  const float* Wg  = (const float*)d_in[6];
  const float* bg  = (const float*)d_in[7];
  float* out_ctx   = (float*)d_out;
  float* out_align = out_ctx + (size_t)Bb * Tt * Hh;

  (void)hipMemsetAsync(d_ws, 0, 256, stream);   // reset grid-barrier state each call
  gmm_attn_lstm<<<dim3(NBLK), dim3(NTHR), 0, stream>>>(
      xin, mem, Wih, Whh, bih, bhh, Wg, bg, out_ctx, out_align, (char*)d_ws);
}
